// MultiheadDiffAttn_71425306132774
// MI455X (gfx1250) — matrix operand order repaired and checked by reading, not compiled, not run on hardware
//
#include <hip/hip_runtime.h>
#include <cstdint>
#include <cstddef>

// ---------------------------------------------------------------------------
// MultiheadDiffAttn for MI455X (gfx1250, wave32, WMMA).
// bf16 convert (weights pre-transposed) -> double-buffered WMMA GEMM with
// async Global->LDS staging -> rotary/relayout -> flash-style dual-softmax
// diff attention (WMMA + async staging) -> WMMA GEMM (Wo).
// ---------------------------------------------------------------------------

typedef __attribute__((ext_vector_type(16))) __bf16 v16bf;
typedef __attribute__((ext_vector_type(8)))  float  v8f;

#define E_DIM   2048
#define NQH     32      // 2*H query heads
#define NH      16      // output heads
#define NKVH    16      // 2*HKV key heads
#define NVH     8       // value heads
#define HD      64      // head dim (per component)
#define DV      128     // value dim (2*HD)
#define T_SEQ   2048
#define NBATCH  2
#define LAMBDA_INIT 0.78360576653f   // 0.8 - 0.6*exp(-0.3*12)

__device__ __forceinline__ uint16_t f32_to_bf16(float f) {
  uint32_t u = __builtin_bit_cast(uint32_t, f);
  uint32_t r = u + 0x7FFFu + ((u >> 16) & 1u);   // round-to-nearest-even
  return (uint16_t)(r >> 16);
}

union FragBF { uint32_t u[8]; v16bf v; };

__device__ __forceinline__ v8f wmma_bf16f32(const FragBF& a, const FragBF& b, v8f c) {
  // D = A(16x32 bf16) * B(32x16 bf16) + C(16x16 f32)
  return __builtin_amdgcn_wmma_f32_16x16x32_bf16(false, a.v, false, b.v,
                                                 (short)0, c, false, false);
}

// K index inside a 32-wide k-step for the 16-bit A-matrix layout (ISA 7.12.2)
__device__ __forceinline__ int a_kidx(int v, int hi) {
  return 2 * v + (v >= 4 ? 8 : 0) + hi * 8;
}
// K index for the 16-bit B-matrix layout (same k order as the A layout)
__device__ __forceinline__ int b_kidx(int v, int hi) {
  return 2 * v + (v >= 4 ? 8 : 0) + hi * 8;
}

// Generic-pointer low 32 bits == wave-relative LDS byte offset (ISA 10.2).
__device__ __forceinline__ uint32_t lds_off(const void* p) {
  return (uint32_t)(uintptr_t)p;
}

// CDNA5 async Global->LDS copy, 16B per lane, tracked by ASYNCcnt.
__device__ __forceinline__ void async_load_b128(uint32_t lds_addr, const void* gaddr) {
  asm volatile("global_load_async_to_lds_b128 %0, %1, off"
               :: "v"(lds_addr), "v"(gaddr) : "memory");
}
__device__ __forceinline__ void wait_async0() {
  asm volatile("s_wait_asynccnt 0x0" ::: "memory");
}

// ---------------------------------------------------------------------------
// Elementwise fp32 -> bf16
// ---------------------------------------------------------------------------
__global__ void cvt_bf16_kernel(const float* __restrict__ src,
                                uint16_t* __restrict__ dst, size_t n) {
  size_t i = (size_t)blockIdx.x * 256 + threadIdx.x;
  if (i < n) dst[i] = f32_to_bf16(src[i]);
}

// fp32 (K x N) -> bf16 transposed (N x K), so GEMM B-staging is a straight copy.
__global__ void cvt_bf16_transpose_kernel(const float* __restrict__ src,
                                          uint16_t* __restrict__ dst,
                                          int K, int N) {
  size_t idx = (size_t)blockIdx.x * 256 + threadIdx.x;
  if (idx >= (size_t)K * N) return;
  int n = (int)(idx % N);
  int k = (int)(idx / N);
  dst[(size_t)n * K + k] = f32_to_bf16(src[idx]);
}

// ---------------------------------------------------------------------------
// GEMM: C(f32, MxN) = A(bf16, MxK, row-major) * Bt(bf16, NxK, row-major)^T
// 64x64 tile / 128 threads (4 waves); each wave a 32x32 quadrant (2x2 WMMA).
// Double-buffered LDS tiles filled with async Global->LDS b128 copies.
// ---------------------------------------------------------------------------
__global__ __launch_bounds__(128) void gemm_bf16_tn(
    const uint16_t* __restrict__ A, const uint16_t* __restrict__ Bt,
    float* __restrict__ C, int M, int N, int K) {
  (void)M;
  __shared__ __align__(16) uint16_t lA[2][64][40];   // [buf][m][k]
  __shared__ __align__(16) uint16_t lB[2][64][40];   // [buf][n][k]

  const int tid = threadIdx.x;
  const int lane = tid & 31, w = tid >> 5;
  const int hi = lane >> 4, ln = lane & 15;
  const int m0 = blockIdx.y * 64;
  const int n0 = blockIdx.x * 64;
  const int m_off = (w >> 1) * 32;
  const int n_off = (w & 1) * 32;

  // staging coords: 2 threads per row, 16 halves (2 x b128) each
  const int srow = tid >> 1;
  const int sc   = (tid & 1) * 16;
  const uint16_t* aPtr = A  + (size_t)(m0 + srow) * K + sc;
  const uint16_t* bPtr = Bt + (size_t)(n0 + srow) * K + sc;
  const uint32_t laOff[2] = { lds_off(&lA[0][srow][sc]), lds_off(&lA[1][srow][sc]) };
  const uint32_t lbOff[2] = { lds_off(&lB[0][srow][sc]), lds_off(&lB[1][srow][sc]) };

  auto stage = [&](int buf, int kt) {
    async_load_b128(laOff[buf],      aPtr + kt);
    async_load_b128(laOff[buf] + 16, aPtr + kt + 8);
    async_load_b128(lbOff[buf],      bPtr + kt);
    async_load_b128(lbOff[buf] + 16, bPtr + kt + 8);
  };

  v8f acc[2][2];
#pragma unroll
  for (int rb = 0; rb < 2; ++rb)
#pragma unroll
    for (int nb = 0; nb < 2; ++nb)
#pragma unroll
      for (int i = 0; i < 8; ++i) acc[rb][nb][i] = 0.0f;

  stage(0, 0);
  wait_async0();
  __syncthreads();

  const int nk = K / 32;
  for (int t = 0; t < nk; ++t) {
    const int buf = t & 1;
    if (t + 1 < nk) stage(buf ^ 1, (t + 1) * 32);   // prefetch next tile

    FragBF af[2], bfg[2];
#pragma unroll
    for (int rb = 0; rb < 2; ++rb)
#pragma unroll
      for (int v = 0; v < 8; ++v)
        af[rb].u[v] = *(const uint32_t*)&lA[buf][m_off + rb * 16 + ln][a_kidx(v, hi)];
#pragma unroll
    for (int nb = 0; nb < 2; ++nb)
#pragma unroll
      for (int v = 0; v < 8; ++v)
        bfg[nb].u[v] = *(const uint32_t*)&lB[buf][n_off + nb * 16 + ln][b_kidx(v, hi)];
#pragma unroll
    for (int rb = 0; rb < 2; ++rb)
#pragma unroll
      for (int nb = 0; nb < 2; ++nb)
        acc[rb][nb] = wmma_bf16f32(af[rb], bfg[nb], acc[rb][nb]);

    wait_async0();       // own async writes for next buffer complete
    __syncthreads();     // all waves done reading cur / writing next
  }

#pragma unroll
  for (int rb = 0; rb < 2; ++rb)
#pragma unroll
    for (int nb = 0; nb < 2; ++nb)
#pragma unroll
      for (int i = 0; i < 8; ++i) {
        int row = m0 + m_off + rb * 16 + i + 8 * hi;
        int col = n0 + n_off + nb * 16 + ln;
        C[(size_t)row * N + col] = acc[rb][nb][i];
      }
}

// ---------------------------------------------------------------------------
// Rotary + scale + relayout for Q: (B,T,2048) f32 -> (B,32,T,64) bf16, *HD^-0.5
// ---------------------------------------------------------------------------
__global__ void rotary_q_kernel(const float* __restrict__ qraw,
                                uint16_t* __restrict__ qh) {
  size_t idx = (size_t)blockIdx.x * 256 + threadIdx.x;  // B*T*32*32 = 2^22
  if (idx >= (size_t)NBATCH * T_SEQ * NQH * 32) return;
  int j    = idx & 31;
  int head = (idx >> 5) & 31;
  int t    = (idx >> 10) & (T_SEQ - 1);
  int b    = (int)(idx >> 21);
  const float* src = qraw + ((size_t)b * T_SEQ + t) * E_DIM + head * HD;
  float x1 = src[j], x2 = src[j + 32];
  float ang = (float)t * __expf(-(float)(2 * j) * (9.210340371976184f / 64.f));
  float s, c;
  __sincosf(ang, &s, &c);
  uint16_t* dst = qh + (((size_t)b * NQH + head) * T_SEQ + t) * HD;
  dst[j]      = f32_to_bf16((x1 * c + x2 * s) * 0.125f);
  dst[j + 32] = f32_to_bf16((-x1 * s + x2 * c) * 0.125f);
}

// Rotary + relayout for K: (B,T,1024) f32 -> (B,16,T,64) bf16
__global__ void rotary_k_kernel(const float* __restrict__ kraw,
                                uint16_t* __restrict__ kh) {
  size_t idx = (size_t)blockIdx.x * 256 + threadIdx.x;  // B*T*16*32 = 2^21
  if (idx >= (size_t)NBATCH * T_SEQ * NKVH * 32) return;
  int j    = idx & 31;
  int head = (idx >> 5) & 15;
  int t    = (idx >> 9) & (T_SEQ - 1);
  int b    = (int)(idx >> 20);
  const float* src = kraw + ((size_t)b * T_SEQ + t) * 1024 + head * HD;
  float x1 = src[j], x2 = src[j + 32];
  float ang = (float)t * __expf(-(float)(2 * j) * (9.210340371976184f / 64.f));
  float s, c;
  __sincosf(ang, &s, &c);
  uint16_t* dst = kh + (((size_t)b * NKVH + head) * T_SEQ + t) * HD;
  dst[j]      = f32_to_bf16(x1 * c + x2 * s);
  dst[j + 32] = f32_to_bf16(-x1 * s + x2 * c);
}

// V relayout (transposed): (B,T,1024) f32 -> (B,8,128,T) bf16
__global__ void pack_v_kernel(const float* __restrict__ vraw,
                              uint16_t* __restrict__ vT) {
  size_t idx = (size_t)blockIdx.x * 256 + threadIdx.x;  // B*T*8*128 = 2^22
  if (idx >= (size_t)NBATCH * T_SEQ * NVH * DV) return;
  int d  = idx & 127;
  int vh = (idx >> 7) & 7;
  int t  = (idx >> 10) & (T_SEQ - 1);
  int b  = (int)(idx >> 21);
  float val = vraw[((size_t)b * T_SEQ + t) * 1024 + vh * DV + d];
  vT[(((size_t)b * NVH + vh) * DV + d) * T_SEQ + t] = f32_to_bf16(val);
}

// lambda = exp(sum q1*k1) - exp(sum q2*k2) + LAMBDA_INIT
__global__ void lambda_kernel(const float* __restrict__ q1, const float* __restrict__ k1,
                              const float* __restrict__ q2, const float* __restrict__ k2,
                              float* __restrict__ lam) {
  __shared__ float s1[64], s2[64];
  int t = threadIdx.x;
  s1[t] = q1[t] * k1[t];
  s2[t] = q2[t] * k2[t];
  __syncthreads();
  for (int off = 32; off >= 1; off >>= 1) {
    if (t < off) { s1[t] += s1[t + off]; s2[t] += s2[t + off]; }
    __syncthreads();
  }
  if (t == 0) *lam = __expf(s1[0]) - __expf(s2[0]) + LAMBDA_INIT;
}

// ---------------------------------------------------------------------------
// Flash-style differential attention.
// grid = (T/64, NH, B), block = 128 (4 waves; wave w owns q-rows w*16..w*16+15)
// ---------------------------------------------------------------------------
__device__ __forceinline__ void softmax_update(v8f* S, float* m, float* l, v8f* acc,
                                               uint16_t (*sPw)[68], int hi, int ln) {
#pragma unroll
  for (int i = 0; i < 8; ++i) {
    float vmax = fmaxf(fmaxf(S[0][i], S[1][i]), fmaxf(S[2][i], S[3][i]));
#pragma unroll
    for (int off = 8; off >= 1; off >>= 1)
      vmax = fmaxf(vmax, __shfl_xor(vmax, off, 16));
    float mnew = fmaxf(m[i], vmax);
    float corr = __expf(m[i] - mnew);
    float rs = 0.f;
#pragma unroll
    for (int kb = 0; kb < 4; ++kb) {
      float p = __expf(S[kb][i] - mnew);
      S[kb][i] = p;
      rs += p;
    }
#pragma unroll
    for (int off = 8; off >= 1; off >>= 1) rs += __shfl_xor(rs, off, 16);
    l[i] = l[i] * corr + rs;
    m[i] = mnew;
#pragma unroll
    for (int db = 0; db < 8; ++db) acc[db][i] *= corr;
#pragma unroll
    for (int kb = 0; kb < 4; ++kb)
      sPw[i + 8 * hi][kb * 16 + ln] = f32_to_bf16(S[kb][i]);
  }
}

__global__ __launch_bounds__(128) void diff_flash_attn_kernel(
    const uint16_t* __restrict__ qh,   // (B, 32, T, 64) bf16, pre-scaled
    const uint16_t* __restrict__ kh,   // (B, 16, T, 64) bf16
    const uint16_t* __restrict__ vT,   // (B, 8, 128, T) bf16
    const float* __restrict__ g,       // (128)
    const float* __restrict__ lamp,    // scalar lambda
    uint16_t* __restrict__ out) {      // (B, T, 2048) bf16
  __shared__ __align__(16) uint16_t sQ0[64][72];
  __shared__ __align__(16) uint16_t sQ1[64][72];
  __shared__ __align__(16) uint16_t sK[64][72];     // [key][dim]
  __shared__ __align__(16) uint16_t sV[128][72];    // [dim][key]
  __shared__ __align__(16) uint16_t sP0[4][16][68];
  __shared__ __align__(16) uint16_t sP1[4][16][68];
  __shared__ float sG[128];

  const int tid = threadIdx.x;
  const int lane = tid & 31, w = tid >> 5;
  const int hi = lane >> 4, ln = lane & 15;
  const int qt = blockIdx.x;
  const int h  = blockIdx.y;
  const int b  = blockIdx.z;
  const int qbase = qt * 64;

  if (tid < 128) sG[tid] = g[tid];

  // ---- async-stage Q0/Q1 tiles (64x64 halves each; 2 threads/row, 32h each)
  const int srow = tid >> 1;
  const int sc   = (tid & 1) * 32;
  const uint16_t* q0g = qh + (((size_t)b * NQH + 2 * h) * T_SEQ + qbase) * HD;
  const uint16_t* q1g = qh + (((size_t)b * NQH + 2 * h + 1) * T_SEQ + qbase) * HD;
  {
    const uint16_t* g0 = q0g + (size_t)srow * HD + sc;
    const uint16_t* g1 = q1g + (size_t)srow * HD + sc;
    uint32_t l0 = lds_off(&sQ0[srow][sc]);
    uint32_t l1 = lds_off(&sQ1[srow][sc]);
#pragma unroll
    for (int c = 0; c < 4; ++c) {
      async_load_b128(l0 + c * 16, g0 + c * 8);
      async_load_b128(l1 + c * 16, g1 + c * 8);
    }
  }
  wait_async0();
  __syncthreads();

  FragBF aQ0[2], aQ1[2];
#pragma unroll
  for (int ks = 0; ks < 2; ++ks)
#pragma unroll
    for (int v = 0; v < 8; ++v) {
      int kk = a_kidx(v, hi) + ks * 32;
      aQ0[ks].u[v] = *(const uint32_t*)&sQ0[w * 16 + ln][kk];
      aQ1[ks].u[v] = *(const uint32_t*)&sQ1[w * 16 + ln][kk];
    }

  float m0r[8], l0r[8], m1r[8], l1r[8];
  v8f acc0[8], acc1[8];
#pragma unroll
  for (int i = 0; i < 8; ++i) {
    m0r[i] = -3.0e38f; l0r[i] = 0.f; m1r[i] = -3.0e38f; l1r[i] = 0.f;
#pragma unroll
    for (int db = 0; db < 8; ++db) { acc0[db][i] = 0.f; acc1[db][i] = 0.f; }
  }

  const uint16_t* kg = kh + ((size_t)b * NKVH + h) * T_SEQ * HD;
  const uint16_t* vg = vT + ((size_t)b * NVH + (h >> 1)) * DV * (size_t)T_SEQ;
  const float lam = *lamp;

  const uint32_t lkOff = lds_off(&sK[srow][sc]);
  const uint32_t lvOff = lds_off(&sV[tid][0]);

  for (int st = 0; st <= qt; ++st) {
    const int sbase = st * 64;
    __syncthreads();   // everyone finished reading previous K/V/P tiles
    {
      const uint16_t* gk = kg + (size_t)(sbase + srow) * HD + sc;
#pragma unroll
      for (int c = 0; c < 4; ++c)
        async_load_b128(lkOff + c * 16, gk + c * 8);
      const uint16_t* gv = vg + (size_t)tid * T_SEQ + sbase;   // row = dim
#pragma unroll
      for (int c = 0; c < 8; ++c)
        async_load_b128(lvOff + c * 16, gv + c * 8);
    }
    wait_async0();
    __syncthreads();

    // S = Q * K^T  (A = 16x64 q rows, B = sK[key][dim] -> [n=key][k=dim])
    v8f S0[4], S1[4];
#pragma unroll
    for (int kb = 0; kb < 4; ++kb) {
      FragBF bk0, bk1;
#pragma unroll
      for (int v = 0; v < 8; ++v) {
        bk0.u[v] = *(const uint32_t*)&sK[kb * 16 + ln][b_kidx(v, hi)];
        bk1.u[v] = *(const uint32_t*)&sK[kb * 16 + ln][b_kidx(v, hi) + 32];
      }
      v8f z;
#pragma unroll
      for (int i = 0; i < 8; ++i) z[i] = 0.f;
      S0[kb] = wmma_bf16f32(aQ0[1], bk1, wmma_bf16f32(aQ0[0], bk0, z));
      S1[kb] = wmma_bf16f32(aQ1[1], bk1, wmma_bf16f32(aQ1[0], bk0, z));
    }

    if (st == qt) {  // causal mask on the diagonal tile
#pragma unroll
      for (int kb = 0; kb < 4; ++kb)
#pragma unroll
        for (int i = 0; i < 8; ++i) {
          int key = sbase + kb * 16 + ln;
          int row = qbase + w * 16 + i + 8 * hi;
          if (key > row) { S0[kb][i] = -3.0e38f; S1[kb][i] = -3.0e38f; }
        }
    }

    softmax_update(S0, m0r, l0r, acc0, sP0[w], hi, ln);
    softmax_update(S1, m1r, l1r, acc1, sP1[w], hi, ln);
    __syncthreads();   // own-wave P stores -> A-frag reads (dscnt + barrier)

    FragBF aP0[2], aP1[2];
#pragma unroll
    for (int ks = 0; ks < 2; ++ks)
#pragma unroll
      for (int v = 0; v < 8; ++v) {
        int kk = a_kidx(v, hi) + ks * 32;
        aP0[ks].u[v] = *(const uint32_t*)&sP0[w][ln][kk];
        aP1[ks].u[v] = *(const uint32_t*)&sP1[w][ln][kk];
      }
#pragma unroll
    for (int db = 0; db < 8; ++db) {
      FragBF bv0, bv1;
#pragma unroll
      for (int v = 0; v < 8; ++v) {
        bv0.u[v] = *(const uint32_t*)&sV[db * 16 + ln][b_kidx(v, hi)];
        bv1.u[v] = *(const uint32_t*)&sV[db * 16 + ln][b_kidx(v, hi) + 32];
      }
      acc0[db] = wmma_bf16f32(aP0[0], bv0, acc0[db]);
      acc0[db] = wmma_bf16f32(aP0[1], bv1, acc0[db]);
      acc1[db] = wmma_bf16f32(aP1[0], bv0, acc1[db]);
      acc1[db] = wmma_bf16f32(aP1[1], bv1, acc1[db]);
    }
  }

  // Epilogue: diff, RMS norm, g, (1-lambda_init), bf16 store.
  float rowscale[8];
#pragma unroll
  for (int i = 0; i < 8; ++i) {
    float inv0 = 1.0f / l0r[i];
    float inv1 = lam / l1r[i];
    float ss = 0.f;
#pragma unroll
    for (int db = 0; db < 8; ++db) {
      float v = acc0[db][i] * inv0 - acc1[db][i] * inv1;
      acc0[db][i] = v;
      ss += v * v;
    }
#pragma unroll
    for (int off = 8; off >= 1; off >>= 1) ss += __shfl_xor(ss, off, 16);
    rowscale[i] = rsqrtf(ss * (1.0f / 128.0f) + 1e-6f) * (1.0f - LAMBDA_INIT);
  }
#pragma unroll
  for (int db = 0; db < 8; ++db)
#pragma unroll
    for (int i = 0; i < 8; ++i) {
      float v = acc0[db][i] * rowscale[i] * sG[db * 16 + ln];
      size_t row = (size_t)b * T_SEQ + qbase + w * 16 + i + 8 * hi;
      out[row * E_DIM + h * DV + db * 16 + ln] = f32_to_bf16(v);
    }
}

// ---------------------------------------------------------------------------
// Host launch
// ---------------------------------------------------------------------------
extern "C" void kernel_launch(void* const* d_in, const int* in_sizes, int n_in,
                              void* d_out, int out_size, void* d_ws, size_t ws_size,
                              hipStream_t stream) {
  (void)in_sizes; (void)n_in; (void)out_size; (void)ws_size;
  const float* x  = (const float*)d_in[0];
  const float* Wq = (const float*)d_in[1];
  const float* Wk = (const float*)d_in[2];
  const float* Wv = (const float*)d_in[3];
  const float* Wo = (const float*)d_in[4];
  const float* lq1 = (const float*)d_in[5];
  const float* lk1 = (const float*)d_in[6];
  const float* lq2 = (const float*)d_in[7];
  const float* lk2 = (const float*)d_in[8];
  const float* g   = (const float*)d_in[9];
  float* out = (float*)d_out;

  char* ws = (char*)d_ws;
  const size_t nBT = (size_t)NBATCH * T_SEQ;             // 4096 rows
  // workspace layout (bytes)
  uint16_t* xb   = (uint16_t*)(ws);                      // x bf16
  uint16_t* Wqt  = (uint16_t*)(ws + 16777216);           // Wq^T bf16 (2048x2048)
  uint16_t* Wkt  = (uint16_t*)(ws + 25165824);           // Wk^T bf16 (1024x2048)
  uint16_t* Wvt  = (uint16_t*)(ws + 29360128);           // Wv^T bf16 (1024x2048)
  uint16_t* Wot  = (uint16_t*)(ws + 33554432);           // Wo^T bf16 (2048x2048)
  float*    qraw = (float*)   (ws + 41943040);
  float*    kraw = (float*)   (ws + 75497472);
  float*    vraw = (float*)   (ws + 92274688);
  uint16_t* qhB  = (uint16_t*)(ws + 109051904);
  uint16_t* khB  = (uint16_t*)(ws + 125829120);
  uint16_t* vTB  = (uint16_t*)(ws + 134217728);
  uint16_t* attn = (uint16_t*)(ws + 142606336);
  float*    lamW = (float*)   (ws + 159383552);

  // 1) convert to bf16 (weights transposed)
  cvt_bf16_kernel<<<(nBT * E_DIM + 255) / 256, 256, 0, stream>>>(x, xb, nBT * E_DIM);
  cvt_bf16_transpose_kernel<<<(4194304 + 255) / 256, 256, 0, stream>>>(Wq, Wqt, 2048, 2048);
  cvt_bf16_transpose_kernel<<<(2097152 + 255) / 256, 256, 0, stream>>>(Wk, Wkt, 2048, 1024);
  cvt_bf16_transpose_kernel<<<(2097152 + 255) / 256, 256, 0, stream>>>(Wv, Wvt, 2048, 1024);
  cvt_bf16_transpose_kernel<<<(4194304 + 255) / 256, 256, 0, stream>>>(Wo, Wot, 2048, 2048);

  // 2) projections (WMMA GEMMs, async double-buffered)
  gemm_bf16_tn<<<dim3(2048 / 64, 4096 / 64), 128, 0, stream>>>(xb, Wqt, qraw, 4096, 2048, 2048);
  gemm_bf16_tn<<<dim3(1024 / 64, 4096 / 64), 128, 0, stream>>>(xb, Wkt, kraw, 4096, 1024, 2048);
  gemm_bf16_tn<<<dim3(1024 / 64, 4096 / 64), 128, 0, stream>>>(xb, Wvt, vraw, 4096, 1024, 2048);

  // 3) rotary + relayout + lambda
  rotary_q_kernel<<<(4194304 + 255) / 256, 256, 0, stream>>>(qraw, qhB);
  rotary_k_kernel<<<(2097152 + 255) / 256, 256, 0, stream>>>(kraw, khB);
  pack_v_kernel  <<<(4194304 + 255) / 256, 256, 0, stream>>>(vraw, vTB);
  lambda_kernel<<<1, 64, 0, stream>>>(lq1, lk1, lq2, lk2, lamW);

  // 4) differential flash attention
  diff_flash_attn_kernel<<<dim3(T_SEQ / 64, NH, NBATCH), 128, 0, stream>>>(
      qhB, khB, vTB, g, lamW, attn);

  // 5) output projection
  gemm_bf16_tn<<<dim3(2048 / 64, 4096 / 64), 128, 0, stream>>>(attn, Wot, out, 4096, 2048, 2048);
}
